// Prot3DGraphModel_84430467105441
// MI455X (gfx1250) — matrix operand order repaired and checked by reading, not compiled, not run on hardware
//
#include <hip/hip_runtime.h>

typedef __attribute__((ext_vector_type(16))) __bf16 v16bf;
typedef __attribute__((ext_vector_type(8)))  __bf16 v8bf;
typedef __attribute__((ext_vector_type(8)))  float  v8f;

#define LEAKY 0.01f

// ---------------- weight transpose + fp32->bf16 : W[K][N] -> Wt[N][K] ----------------
__global__ void k_wprep(const float* __restrict__ W, __bf16* __restrict__ Wt, int K, int N) {
  int i = blockIdx.x * blockDim.x + threadIdx.x;
  if (i >= K * N) return;
  int k = i / N, n = i % N;
  Wt[(size_t)n * K + k] = (__bf16)W[(size_t)k * N + n];
}

// ---------------- node projection: concat(embed[seq], node_s) @ W + b -> bf16 [N,128] ----
__global__ void k_node_proj(const int* __restrict__ seq, const float* __restrict__ node_s,
                            const float* __restrict__ emb, const float* __restrict__ Wn,
                            const float* __restrict__ bn, __bf16* __restrict__ xb, int Nn) {
  int i = blockIdx.x * blockDim.x + threadIdx.x;
  if (i >= Nn * 128) return;
  int n = i >> 7, j = i & 127;
  const float* er = emb + (size_t)seq[n] * 20;
  float acc = bn[j];
#pragma unroll
  for (int c = 0; c < 20; ++c) acc += er[c] * Wn[c * 128 + j];
  const float* ns = node_s + (size_t)n * 6;
#pragma unroll
  for (int c = 0; c < 6; ++c) acc += ns[c] * Wn[(20 + c) * 128 + j];
  xb[(size_t)n * 128 + j] = (__bf16)acc;
}

// ---------------- edge projection: edge_s @ W + b -> bf16 [E,128] ----------------
__global__ void k_edge_proj(const float* __restrict__ es, const float* __restrict__ We,
                            const float* __restrict__ be, __bf16* __restrict__ eb, int E) {
  long long i = (long long)blockIdx.x * blockDim.x + threadIdx.x;
  if (i >= (long long)E * 128) return;
  int e = (int)(i >> 7), j = (int)(i & 127);
  const float* r = es + (size_t)e * 39;
  float acc = be[j];
#pragma unroll
  for (int c = 0; c < 39; ++c) acc += r[c] * We[c * 128 + j];
  eb[(size_t)e * 128 + j] = (__bf16)acc;
}

// ---------------- WMMA GEMM: C[M,N] = A[M,K](bf16) x Wt[N,K](bf16)^T + bias ----------------
// One 16x16 output tile per wave32. M%16==0, N%16==0, K%32==0.
__global__ void k_gemm(const __bf16* __restrict__ A, const __bf16* __restrict__ Wt,
                       const float* __restrict__ bias, void* __restrict__ Cout,
                       int obf, int M, int N, int K) {
  int wid  = (int)(((unsigned)blockIdx.x * blockDim.x + threadIdx.x) >> 5);
  int lane = threadIdx.x & 31;
  int ntiles = N >> 4;
  int tiles  = (M >> 4) * ntiles;
  if (wid >= tiles) return;              // wave-uniform: EXEC stays all-ones for WMMA
  int m0 = (wid / ntiles) << 4;
  int n0 = (wid % ntiles) << 4;
  int kh = lane >> 4, l15 = lane & 15;
  // A 16-bit 16x32 fragment: lane m=l15, j<8 -> K=kh*8+j ; j>=8 -> K=16+kh*8+(j-8)
  const __bf16* ar = A  + (size_t)(m0 + l15) * K + kh * 8;
  // B 16-bit 32x16 fragment: lane col n=l15, j<8 -> K=kh*8+j ; j>=8 -> K=16+kh*8+(j-8)
  const __bf16* br = Wt + (size_t)(n0 + l15) * K + kh * 8;
  v8f acc = {};
  for (int kk = 0; kk < K; kk += 32) {
    v8bf alo = *(const v8bf*)(ar + kk);
    v8bf ahi = *(const v8bf*)(ar + kk + 16);
    v8bf blo = *(const v8bf*)(br + kk);
    v8bf bhi = *(const v8bf*)(br + kk + 16);
    v16bf a, b;
#pragma unroll
    for (int j = 0; j < 8; ++j) { a[j] = alo[j]; a[j + 8] = ahi[j];
                                  b[j] = blo[j]; b[j + 8] = bhi[j]; }
    acc = __builtin_amdgcn_wmma_f32_16x16x32_bf16(false, a, false, b, (short)0, acc,
                                                  false, false);
  }
  float bv = bias ? bias[n0 + l15] : 0.0f;
  // C/D layout: VGPR r -> row m0 + kh*8 + r, col n0 + l15
  size_t base = (size_t)(m0 + kh * 8) * N + n0 + l15;
  if (obf) {
    __bf16* c = (__bf16*)Cout;
#pragma unroll
    for (int r = 0; r < 8; ++r) c[base + (size_t)r * N] = (__bf16)(acc[r] + bv);
  } else {
    float* c = (float*)Cout;
#pragma unroll
    for (int r = 0; r < 8; ++r) c[base + (size_t)r * N] = acc[r] + bv;
  }
}

// ---- monotone float<->uint map for atomicMax-based segment max ----
__device__ __forceinline__ unsigned fmap(float f) {
  unsigned u = __float_as_uint(f);
  return (u & 0x80000000u) ? ~u : (u | 0x80000000u);
}
__device__ __forceinline__ float funmap(unsigned u) {
  return __uint_as_float((u & 0x80000000u) ? (u & 0x7fffffffu) : ~u);
}

// ---------------- alpha = dot(q[dst], k[src]+e_lin)/sqrt(D); segment max ----------------
__global__ void k_alpha(const float* __restrict__ q, const float* __restrict__ kk,
                        const __bf16* __restrict__ elin, const int* __restrict__ src,
                        const int* __restrict__ dst, float* __restrict__ alpha,
                        unsigned* __restrict__ mmax, int E, int D) {
  int wid  = (int)(((unsigned)blockIdx.x * blockDim.x + threadIdx.x) >> 5);
  int lane = threadIdx.x & 31;
  if (wid >= E) return;
  int s = src[wid], d = dst[wid];
  const float*  qr = q  + (size_t)d * D;
  const float*  kr = kk + (size_t)s * D;
  const __bf16* er = elin + (size_t)wid * D;
  float p = 0.0f;
  for (int j = lane; j < D; j += 32) p += qr[j] * (kr[j] + (float)er[j]);
  for (int off = 16; off; off >>= 1) p += __shfl_xor(p, off, 32);
  if (lane == 0) {
    float a = p * rsqrtf((float)D);
    alpha[wid] = a;
    atomicMax(&mmax[d], fmap(a));
  }
}

// ---------------- a = exp(alpha - m[dst]); denom[dst] += a ----------------
__global__ void k_soft(const float* __restrict__ alpha, const unsigned* __restrict__ mmax,
                       const int* __restrict__ dst, float* __restrict__ denom,
                       float* __restrict__ aexp, int E) {
  int i = blockIdx.x * blockDim.x + threadIdx.x;
  if (i >= E) return;
  int d = dst[i];
  float a = __expf(alpha[i] - funmap(mmax[d]));
  aexp[i] = a;
  atomicAdd(&denom[d], a);
}

// ---------------- num[dst] += a * (v[src] + e_lin) ----------------
__global__ void k_agg(const float* __restrict__ v, const __bf16* __restrict__ elin,
                      const int* __restrict__ src, const int* __restrict__ dst,
                      const float* __restrict__ aexp, float* __restrict__ num,
                      int E, int D) {
  int wid  = (int)(((unsigned)blockIdx.x * blockDim.x + threadIdx.x) >> 5);
  int lane = threadIdx.x & 31;
  if (wid >= E) return;
  int s = src[wid], d = dst[wid];
  float a = aexp[wid];
  const float*  vr = v + (size_t)s * D;
  const __bf16* er = elin + (size_t)wid * D;
  float* nr = num + (size_t)d * D;
  for (int j = lane; j < D; j += 32) atomicAdd(&nr[j], a * (vr[j] + (float)er[j]));
}

// ---------------- x = leaky_relu(num/denom + skip); emit fp32 + bf16 ----------------
__global__ void k_finish(const float* __restrict__ num, const float* __restrict__ denom,
                         const float* __restrict__ skip, float* __restrict__ xf,
                         __bf16* __restrict__ xb, int Nn, int D) {
  int i = blockIdx.x * blockDim.x + threadIdx.x;
  if (i >= Nn * D) return;
  int n = i / D;
  float dn  = denom[n];
  float agg = dn > 0.0f ? num[i] / dn : 0.0f;
  float val = agg + skip[i];
  val = val > 0.0f ? val : LEAKY * val;
  xf[i] = val;
  xb[i] = (__bf16)val;
}

// ---------------- global mean pool ----------------
__global__ void k_count(const int* __restrict__ batch, float* __restrict__ counts, int Nn) {
  int i = blockIdx.x * blockDim.x + threadIdx.x;
  if (i >= Nn) return;
  atomicAdd(&counts[batch[i]], 1.0f);
}
__global__ void k_pool(const float* __restrict__ xf, const int* __restrict__ batch,
                       float* __restrict__ pooled, int Nn, int D) {
  int i = blockIdx.x * blockDim.x + threadIdx.x;
  if (i >= Nn * D) return;
  int n = i / D, j = i % D;
  atomicAdd(&pooled[(size_t)batch[n] * D + j], xf[i]);
}
__global__ void k_poolfin(const float* __restrict__ pooled, const float* __restrict__ counts,
                          float* __restrict__ out, int G, int D) {
  int i = blockIdx.x * blockDim.x + threadIdx.x;
  if (i >= G * D) return;
  out[i] = pooled[i] / fmaxf(counts[i / D], 1.0f);
}

extern "C" void kernel_launch(void* const* d_in, const int* in_sizes, int n_in,
                              void* d_out, int out_size, void* d_ws, size_t ws_size,
                              hipStream_t stream) {
  (void)n_in; (void)out_size; (void)ws_size;
  const int Nn = in_sizes[0];
  const int E  = in_sizes[3] / 2;

  const int*   seq    = (const int*)  d_in[0];
  const float* node_s = (const float*)d_in[1];
  const float* edge_s = (const float*)d_in[2];
  const int*   src    = (const int*)  d_in[3];
  const int*   dst    = src + E;
  const int*   batch  = (const int*)  d_in[4];
  const float* emb    = (const float*)d_in[5];
  const float* pnw    = (const float*)d_in[6];
  const float* pnb    = (const float*)d_in[7];
  const float* pew    = (const float*)d_in[8];
  const float* peb    = (const float*)d_in[9];

  const int DIMS[3][2] = {{128,128},{128,256},{256,256}};

  // ---- workspace bump allocator (256B aligned) ----
  char* w = (char*)d_ws;
  size_t off = 0;
  auto alloc = [&](size_t bytes) -> void* {
    void* p = w + off; off = (off + bytes + 255) & ~(size_t)255; return p;
  };
  __bf16*  xb    = (__bf16*) alloc((size_t)Nn * 256 * 2);
  float*   xf    = (float*)  alloc((size_t)Nn * 256 * 4);
  __bf16*  ebf   = (__bf16*) alloc((size_t)E  * 128 * 2);
  __bf16*  elin  = (__bf16*) alloc((size_t)E  * 256 * 2);
  float*   qf    = (float*)  alloc((size_t)Nn * 256 * 4);
  float*   kf    = (float*)  alloc((size_t)Nn * 256 * 4);
  float*   vf    = (float*)  alloc((size_t)Nn * 256 * 4);
  float*   sf    = (float*)  alloc((size_t)Nn * 256 * 4);
  float*   num   = (float*)  alloc((size_t)Nn * 256 * 4);
  float*   alpha = (float*)  alloc((size_t)E * 4);
  float*   aexp  = (float*)  alloc((size_t)E * 4);
  unsigned* mmax = (unsigned*)alloc((size_t)Nn * 4);
  float*   denom = (float*)  alloc((size_t)Nn * 4);
  float*   pooled= (float*)  alloc((size_t)16 * 256 * 4);
  float*   counts= (float*)  alloc((size_t)16 * 4);
  __bf16* Wt[3][5];
  for (int l = 0; l < 3; ++l) {
    int din = DIMS[l][0], dout = DIMS[l][1];
    Wt[l][0] = (__bf16*)alloc((size_t)din * dout * 2);  // q
    Wt[l][1] = (__bf16*)alloc((size_t)din * dout * 2);  // k
    Wt[l][2] = (__bf16*)alloc((size_t)din * dout * 2);  // v
    Wt[l][3] = (__bf16*)alloc((size_t)128 * dout * 2);  // edge
    Wt[l][4] = (__bf16*)alloc((size_t)din * dout * 2);  // skip
  }

  auto grid1 = [](long long n, int b) { return dim3((unsigned)((n + b - 1) / b)); };

  // ---- weight prep (transpose + bf16) ----
  for (int l = 0; l < 3; ++l) {
    int din = DIMS[l][0], dout = DIMS[l][1];
    const float* Wq = (const float*)d_in[10 + 9 * l + 0];
    const float* Wk = (const float*)d_in[10 + 9 * l + 2];
    const float* Wv = (const float*)d_in[10 + 9 * l + 4];
    const float* We = (const float*)d_in[10 + 9 * l + 6];
    const float* Ws = (const float*)d_in[10 + 9 * l + 7];
    k_wprep<<<grid1((long long)din * dout, 256), 256, 0, stream>>>(Wq, Wt[l][0], din, dout);
    k_wprep<<<grid1((long long)din * dout, 256), 256, 0, stream>>>(Wk, Wt[l][1], din, dout);
    k_wprep<<<grid1((long long)din * dout, 256), 256, 0, stream>>>(Wv, Wt[l][2], din, dout);
    k_wprep<<<grid1((long long)128 * dout, 256), 256, 0, stream>>>(We, Wt[l][3], 128, dout);
    k_wprep<<<grid1((long long)din * dout, 256), 256, 0, stream>>>(Ws, Wt[l][4], din, dout);
  }

  // ---- input projections ----
  k_node_proj<<<grid1((long long)Nn * 128, 256), 256, 0, stream>>>(seq, node_s, emb, pnw, pnb, xb, Nn);
  k_edge_proj<<<grid1((long long)E * 128, 256), 256, 0, stream>>>(edge_s, pew, peb, ebf, E);

  // ---- layers ----
  for (int l = 0; l < 3; ++l) {
    int din = DIMS[l][0], dout = DIMS[l][1];
    const float* bq = (const float*)d_in[10 + 9 * l + 1];
    const float* bk = (const float*)d_in[10 + 9 * l + 3];
    const float* bv = (const float*)d_in[10 + 9 * l + 5];
    const float* bs = (const float*)d_in[10 + 9 * l + 8];

    auto gemm = [&](const __bf16* A, const __bf16* Wtp, const float* bias,
                    void* C, int obf, int M, int Nc, int K) {
      long long waves = (long long)(M / 16) * (Nc / 16);
      k_gemm<<<grid1(waves * 32, 256), 256, 0, stream>>>(A, Wtp, bias, C, obf, M, Nc, K);
    };
    gemm(xb,  Wt[l][0], bq,      qf,   0, Nn, dout, din);
    gemm(xb,  Wt[l][1], bk,      kf,   0, Nn, dout, din);
    gemm(xb,  Wt[l][2], bv,      vf,   0, Nn, dout, din);
    gemm(xb,  Wt[l][4], bs,      sf,   0, Nn, dout, din);
    gemm(ebf, Wt[l][3], nullptr, elin, 1, E,  dout, 128);   // big edge GEMM, bf16 out

    hipMemsetAsync(mmax,  0, (size_t)Nn * 4, stream);
    hipMemsetAsync(denom, 0, (size_t)Nn * 4, stream);
    hipMemsetAsync(num,   0, (size_t)Nn * dout * 4, stream);

    k_alpha<<<grid1((long long)E * 32, 256), 256, 0, stream>>>(qf, kf, elin, src, dst,
                                                               alpha, mmax, E, dout);
    k_soft<<<grid1(E, 256), 256, 0, stream>>>(alpha, mmax, dst, denom, aexp, E);
    k_agg<<<grid1((long long)E * 32, 256), 256, 0, stream>>>(vf, elin, src, dst, aexp,
                                                             num, E, dout);
    k_finish<<<grid1((long long)Nn * dout, 256), 256, 0, stream>>>(num, denom, sf, xf, xb,
                                                                   Nn, dout);
  }

  // ---- global mean pool ----
  hipMemsetAsync(pooled, 0, (size_t)16 * 256 * 4, stream);
  hipMemsetAsync(counts, 0, (size_t)16 * 4, stream);
  k_count<<<grid1(Nn, 256), 256, 0, stream>>>(batch, counts, Nn);
  k_pool<<<grid1((long long)Nn * 256, 256), 256, 0, stream>>>(xf, batch, pooled, Nn, 256);
  k_poolfin<<<grid1(16 * 256, 256), 256, 0, stream>>>(pooled, counts, (float*)d_out, 16, 256);
}
